// STDF_22995254903529
// MI455X (gfx1250) — matrix operand order repaired and checked by reading, not compiled, not run on hardware
//
#include <hip/hip_runtime.h>
#include <hip/hip_bf16.h>

typedef __attribute__((ext_vector_type(16))) _Float16 v16h;
typedef __attribute__((ext_vector_type(8)))  float    v8f;

#define MODE_CONV  0
#define MODE_CONVT 1

struct ConvParams {
  const float* in;
  const float* w;
  const float* bias;
  float*       out;
  int B;
  int Cin, CinTotal, cinOff;     // input channels used / buffer channel count / first channel
  int Din, Hin, Win;
  int Cout, CoutTotal, coutOff;  // output channels / buffer channel count / first channel
  int Dout, Hout, Wout;
  int stride, pad, padD;
  int relu;
  int wOstride;                  // MODE_CONVT: O-dim stride inside (I,O,4,4) weights
};

// ---- B-operand gather (im2col on the fly), branchless: clamp index, always
// load (address is always in-bounds), then select 0 for invalid taps. ----
template <int MODE, int KS, int KD>
__device__ __forceinline__ float gather_in(const ConvParams& p, int k, int Ktot,
                                           bool nvalid, int ob, int od, int oy, int ox) {
  int kc = min(k, Ktot - 1);
  if constexpr (MODE == MODE_CONV) {
    int kx = kc % KS;
    int t2 = kc / KS;
    int ky = t2 % KS; t2 /= KS;
    int kd = t2 % KD;
    int ci = t2 / KD;
    int iy = oy * p.stride - p.pad + ky;
    int ix = ox * p.stride - p.pad + kx;
    int iz = od - p.padD + kd;
    bool ok = nvalid && (k < Ktot) &&
              iy >= 0 && iy < p.Hin && ix >= 0 && ix < p.Win &&
              iz >= 0 && iz < p.Din;
    int iyc = min(max(iy, 0), p.Hin - 1);
    int ixc = min(max(ix, 0), p.Win - 1);
    int izc = min(max(iz, 0), p.Din - 1);
    float v = p.in[(((ob * p.CinTotal + p.cinOff + ci) * p.Din + izc) * p.Hin + iyc) * p.Win + ixc];
    return ok ? v : 0.f;
  } else {
    // transposed conv 4x4, stride 2, pad 1: oy = 2*iy - 1 + ky
    int kx = kc & 3, ky = (kc >> 2) & 3, ci = kc >> 4;
    int ty = oy + 1 - ky, tx = ox + 1 - kx;
    int iy = ty >> 1, ix = tx >> 1;
    bool ok = nvalid && (k < Ktot) &&
              ty >= 0 && tx >= 0 && !(ty & 1) && !(tx & 1) &&
              iy < p.Hin && ix < p.Win;
    int iyc = min(max(iy, 0), p.Hin - 1);
    int ixc = min(max(ix, 0), p.Win - 1);
    float v = p.in[((ob * p.CinTotal + p.cinOff + ci) * p.Hin + iyc) * p.Win + ixc];
    return ok ? v : 0.f;
  }
}

// Generic implicit-GEMM conv on V_WMMA_F32_16X16X32_F16.
// Block = 128 threads = 4 waves. Weights for this block's 32 out-channels are
// staged once into LDS as f16 (zero-padded to a 32-multiple of K). Each wave
// computes a 32(M) x 32(N) register tile: 2 A fragments (LDS) x 2 B fragments
// (global gather) -> 4 WMMAs per K-chunk, amortizing both operands.
// A fragment: lane&15 = M row; K halves: lanes<16 -> {0..15}, lanes>=16 -> {16..31}.
// B fragment: lane&15 = N col; K halves: lanes<16 -> {0..15}, lanes>=16 -> {16..31}.
template <int MODE, int KS, int KD>
__global__ __launch_bounds__(128)
void conv_wmma(ConvParams p) {
  extern __shared__ _Float16 wlds[];
  const int tid    = threadIdx.x;
  const int lane   = tid & 31;
  const int waveId = tid >> 5;
  const int l15    = lane & 15;
  const int hiHalf = lane >> 4;
  const int m0     = blockIdx.y << 5;      // 32 output channels per block
  const int HWo    = p.Hout * p.Wout;
  const int Npix   = p.B * p.Dout * HWo;
  const int Ktot   = p.Cin * KD * KS * KS;
  const int KtotUp = (Ktot + 31) & ~31;
  const int ldk    = KtotUp + 8;           // bank-decorrelation pad (halves)

  // ---- stage weight tile (32 x KtotUp f16, zero-padded) into LDS ----
  for (int r = 0; r < 32; ++r) {
    int m  = m0 + r;
    int mc = min(m, p.Cout - 1);
    for (int k = tid; k < KtotUp; k += 128) {
      int kc = min(k, Ktot - 1);
      float wv;
      if constexpr (MODE == MODE_CONV) {
        wv = p.w[mc * Ktot + kc];          // O x (Cin*KD*KS*KS) contiguous
      } else {
        int kx = kc & 3, ky = (kc >> 2) & 3, ci = kc >> 4;
        wv = p.w[((ci * p.wOstride + mc) << 4) + (ky << 2) + kx];
      }
      bool ok = (m < p.Cout) && (k < Ktot);
      wlds[r * ldk + k] = ok ? (_Float16)wv : (_Float16)0.f;
    }
  }
  __syncthreads();

  // ---- two N tiles (16 pixels each) per wave ----
  const int nBase = (blockIdx.x * 8 + waveId * 2) << 4;
  int nc0 = nBase + l15;
  int nc1 = nc0 + 16;
  bool v0 = nc0 < Npix, v1 = nc1 < Npix;
  int pix0 = v0 ? nc0 : 0, pix1 = v1 ? nc1 : 0;
  int ox0 = pix0 % p.Wout; int t = pix0 / p.Wout;
  int oy0 = t % p.Hout; t /= p.Hout;
  int od0 = t % p.Dout; int ob0 = t / p.Dout;
  int ox1 = pix1 % p.Wout; t = pix1 / p.Wout;
  int oy1 = t % p.Hout; t /= p.Hout;
  int od1 = t % p.Dout; int ob1 = t / p.Dout;

  v8f acc00 = {0.f,0.f,0.f,0.f,0.f,0.f,0.f,0.f};
  v8f acc01 = acc00, acc10 = acc00, acc11 = acc00;

  const _Float16* wrow = wlds + l15 * ldk;

  for (int kk = 0; kk < KtotUp; kk += 32) {
    v16h a0, a1, b0, b1;
#pragma unroll
    for (int h = 0; h < 16; ++h) {
      int kA = kk + h + (hiHalf << 4);
      a0[h] = wrow[kA];                     // M rows m0..m0+15
      a1[h] = wrow[(ldk << 4) + kA];        // M rows m0+16..m0+31
      int kB = kk + h + (hiHalf << 4);
      b0[h] = (_Float16)gather_in<MODE, KS, KD>(p, kB, Ktot, v0, ob0, od0, oy0, ox0);
      b1[h] = (_Float16)gather_in<MODE, KS, KD>(p, kB, Ktot, v1, ob1, od1, oy1, ox1);
    }
    acc00 = __builtin_amdgcn_wmma_f32_16x16x32_f16(false, a0, false, b0, (short)0, acc00, false, false);
    acc01 = __builtin_amdgcn_wmma_f32_16x16x32_f16(false, a0, false, b1, (short)0, acc01, false, false);
    acc10 = __builtin_amdgcn_wmma_f32_16x16x32_f16(false, a1, false, b0, (short)0, acc10, false, false);
    acc11 = __builtin_amdgcn_wmma_f32_16x16x32_f16(false, a1, false, b1, (short)0, acc11, false, false);
  }

  // Epilogue: C/D layout -> VGPR r, lane<16: (M=r, N=lane), lane>=16: (M=r+8, N=lane-16)
#pragma unroll
  for (int r = 0; r < 8; ++r) {
    int ma = m0 + r + (hiHalf << 3);        // tile 0 rows
    int mb = ma + 16;                       // tile 1 rows
    if (ma < p.Cout) {
      float bv = p.bias[ma];
      if (v0) {
        float v = acc00[r] + bv;
        if (p.relu) v = fmaxf(v, 0.f);
        p.out[(((ob0 * p.CoutTotal + p.coutOff + ma) * p.Dout + od0) * p.Hout + oy0) * p.Wout + ox0] = v;
      }
      if (v1) {
        float v = acc01[r] + bv;
        if (p.relu) v = fmaxf(v, 0.f);
        p.out[(((ob1 * p.CoutTotal + p.coutOff + ma) * p.Dout + od1) * p.Hout + oy1) * p.Wout + ox1] = v;
      }
    }
    if (mb < p.Cout) {
      float bv = p.bias[mb];
      if (v0) {
        float v = acc10[r] + bv;
        if (p.relu) v = fmaxf(v, 0.f);
        p.out[(((ob0 * p.CoutTotal + p.coutOff + mb) * p.Dout + od0) * p.Hout + oy0) * p.Wout + ox0] = v;
      }
      if (v1) {
        float v = acc11[r] + bv;
        if (p.relu) v = fmaxf(v, 0.f);
        p.out[(((ob1 * p.CoutTotal + p.coutOff + mb) * p.Dout + od1) * p.Hout + oy1) * p.Wout + ox1] = v;
      }
    }
  }
}

__global__ void add_relu_k(const float* a, const float* b, float* o, int n) {
  int i = blockIdx.x * blockDim.x + threadIdx.x;
  if (i < n) o[i] = fmaxf(a[i] + b[i], 0.f);
}

__global__ void mean_d_k(const float* in, float* out, int BC, int D, int HW) {
  int i = blockIdx.x * blockDim.x + threadIdx.x;
  if (i >= BC * HW) return;
  int bc = i / HW, px = i % HW;
  float s = 0.f;
  for (int d = 0; d < D; ++d) s += in[(bc * D + d) * HW + px];
  out[i] = s / (float)D;
}

// Deformable-conv bilinear sampling with sigmoid mask.
__global__ void deform_sample_k(const float* __restrict__ x,
                                const float* __restrict__ om,
                                float* __restrict__ smp,
                                int B, int C, int H, int W) {
  const int K = 9, HW = H * W;
  int i = blockIdx.x * blockDim.x + threadIdx.x;
  int total = B * C * K * HW;
  if (i >= total) return;
  int px = i % W;  int t = i / W;
  int py_ = t % H; t /= H;
  int k = t % K;   t /= K;
  int c = t % C;   int b = t / C;

  const int OMC = C * 3 * K; // 189
  const float* omb = om + (size_t)b * OMC * HW;
  int sp = py_ * W + px;
  float dy = omb[(((c * K + k) * 2 + 0) * HW) + sp];
  float dx = omb[(((c * K + k) * 2 + 1) * HW) + sp];
  float mk = omb[((C * 2 * K + c * K + k) * HW) + sp];
  mk = 1.f / (1.f + expf(-mk));

  float py = (float)py_ + (float)(k / 3 - 1) + dy;
  float qx = (float)px  + (float)(k % 3 - 1) + dx;
  float fy = floorf(py), fx = floorf(qx);
  int y0 = (int)fy, x0 = (int)fx;
  float wy1 = py - fy, wx1 = qx - fx;
  float wy0 = 1.f - wy1, wx0 = 1.f - wx1;

  const float* img = x + ((size_t)b * C + c) * HW;
  auto corner = [&](int yy, int xx) -> float {
    bool ok = yy >= 0 && yy < H && xx >= 0 && xx < W;
    int yc = min(max(yy, 0), H - 1);
    int xc = min(max(xx, 0), W - 1);
    float v = img[yc * W + xc];
    return ok ? v : 0.f;
  };
  float v = wy0 * wx0 * corner(y0, x0)     + wy0 * wx1 * corner(y0, x0 + 1)
          + wy1 * wx0 * corner(y0 + 1, x0) + wy1 * wx1 * corner(y0 + 1, x0 + 1);
  smp[((size_t)(b * C * K) + c * K + k) * HW + sp] = v * mk;
}

template <int MODE, int KS, int KD>
static void launch_conv(hipStream_t s,
                        const float* in, const float* w, const float* bias, float* out,
                        int B, int Cin, int CinTotal, int cinOff,
                        int Din, int Hin, int Win,
                        int Cout, int CoutTotal, int coutOff,
                        int Dout, int Hout, int Wout,
                        int stride, int pad, int padD,
                        int relu, int wOstride) {
  ConvParams p;
  p.in = in; p.w = w; p.bias = bias; p.out = out;
  p.B = B; p.Cin = Cin; p.CinTotal = CinTotal; p.cinOff = cinOff;
  p.Din = Din; p.Hin = Hin; p.Win = Win;
  p.Cout = Cout; p.CoutTotal = CoutTotal; p.coutOff = coutOff;
  p.Dout = Dout; p.Hout = Hout; p.Wout = Wout;
  p.stride = stride; p.pad = pad; p.padD = padD;
  p.relu = relu; p.wOstride = wOstride;
  int Npix = B * Dout * Hout * Wout;
  int Ktot = Cin * KD * KS * KS;
  int KtotUp = (Ktot + 31) & ~31;
  size_t shmem = (size_t)32 * (KtotUp + 8) * sizeof(_Float16);
  dim3 grid((Npix + 127) / 128, (Cout + 31) / 32);
  conv_wmma<MODE, KS, KD><<<grid, 128, shmem, s>>>(p);
}

extern "C" void kernel_launch(void* const* d_in, const int* in_sizes, int n_in,
                              void* d_out, int out_size, void* d_ws, size_t ws_size,
                              hipStream_t stream) {
  (void)in_sizes; (void)n_in; (void)out_size; (void)ws_size;
  const float* inp = (const float*)d_in[0];
  auto I = [&](int i) { return (const float*)d_in[i]; };

  const int B = 2, C = 7, H = 160, W = 160, nf = 32;
  const int HW = H * W, H2 = 80, W2 = 80, H4 = 40, W4 = 40, H8 = 20, W8 = 20;

  float* ws = (float*)d_ws;
  size_t off = 0;
  auto carve = [&](size_t n) { float* r = ws + off; off += n; return r; };
  float* f0    = carve((size_t)B * nf * HW);
  float* dn1a  = carve((size_t)B * nf * H2 * W2);
  float* cat1  = carve((size_t)B * 2 * nf * H2 * W2); // [convT ch0-31 | f1 ch32-63]
  float* dn2a  = carve((size_t)B * nf * H4 * W4);
  float* cat2  = carve((size_t)B * 2 * nf * H4 * W4); // [convT ch0-31 | f2 ch32-63]
  float* t20a  = carve((size_t)B * nf * H8 * W8);
  float* t20b  = carve((size_t)B * nf * H8 * W8);
  float* up2a  = carve((size_t)B * nf * H4 * W4);
  float* up1a  = carve((size_t)B * nf * H2 * W2);
  float* tfull = carve((size_t)B * nf * HW);
  float* xmean = carve((size_t)B * nf * HW);
  float* tfbuf = carve((size_t)B * nf * HW);
  float* fused = carve((size_t)B * nf * HW);
  float* woutb = carve((size_t)B * nf * HW);
  float* x3a   = carve((size_t)B * nf * 7 * HW);
  float* x3b   = carve((size_t)B * nf * 7 * HW);
  float* om      = x3a; // reuse after 3D path done (needs 9,676,800 <= 11,468,800)
  float* sampled = x3b; // reuse (needs 3,225,600)

  // ---- encoder ----
  launch_conv<MODE_CONV,3,1>(stream, inp,  I(1),  I(2),  f0,   B, C,  C,  0, 1, H,  W,  nf, nf, 0, 1, H,  W,  1, 1, 0, 1, 0);
  launch_conv<MODE_CONV,3,1>(stream, f0,   I(3),  I(4),  dn1a, B, nf, nf, 0, 1, H,  W,  nf, nf, 0, 1, H2, W2, 2, 1, 0, 1, 0);
  launch_conv<MODE_CONV,3,1>(stream, dn1a, I(5),  I(6),  cat1, B, nf, nf, 0, 1, H2, W2, nf, 2*nf, nf, 1, H2, W2, 1, 1, 0, 1, 0); // f1 -> cat1[32:]
  launch_conv<MODE_CONV,3,1>(stream, cat1, I(11), I(12), dn2a, B, nf, 2*nf, nf, 1, H2, W2, nf, nf, 0, 1, H4, W4, 2, 1, 0, 1, 0);
  launch_conv<MODE_CONV,3,1>(stream, dn2a, I(13), I(14), cat2, B, nf, nf, 0, 1, H4, W4, nf, 2*nf, nf, 1, H4, W4, 1, 1, 0, 1, 0); // f2 -> cat2[32:]

  // ---- bottleneck ----
  launch_conv<MODE_CONV,3,1>(stream, cat2, I(19), I(20), t20a, B, nf, 2*nf, nf, 1, H4, W4, nf, nf, 0, 1, H8, W8, 2, 1, 0, 1, 0);
  launch_conv<MODE_CONV,3,1>(stream, t20a, I(21), I(22), t20b, B, nf, nf, 0, 1, H8, W8, nf, nf, 0, 1, H8, W8, 1, 1, 0, 1, 0);
  launch_conv<MODE_CONVT,4,1>(stream, t20b, I(23), I(24), cat2, B, nf, nf, 0, 1, H8, W8, nf, 2*nf, 0, 1, H4, W4, 2, 1, 0, 1, nf); // -> cat2[:32]

  // ---- decoder ----
  launch_conv<MODE_CONV,3,1>(stream, cat2, I(15), I(16), up2a, B, 2*nf, 2*nf, 0, 1, H4, W4, nf, nf, 0, 1, H4, W4, 1, 1, 0, 1, 0);
  launch_conv<MODE_CONVT,4,1>(stream, up2a, I(17), I(18), cat1, B, nf, nf, 0, 1, H4, W4, nf, 2*nf, 0, 1, H2, W2, 2, 1, 0, 1, nf); // -> cat1[:32]
  launch_conv<MODE_CONV,3,1>(stream, cat1, I(7),  I(8),  up1a, B, 2*nf, 2*nf, 0, 1, H2, W2, nf, nf, 0, 1, H2, W2, 1, 1, 0, 1, 0);
  launch_conv<MODE_CONVT,4,1>(stream, up1a, I(9),  I(10), tfull,B, nf, nf, 0, 1, H2, W2, nf, nf, 0, 1, H,  W,  2, 1, 0, 1, nf);

  // ---- temporal 3D branch ----
  launch_conv<MODE_CONV,3,3>(stream, inp, I(27), I(28), x3a, B, 1,  1,  0, 7, H, W, nf, nf, 0, 7, H, W, 1, 1, 1, 1, 0);
  launch_conv<MODE_CONV,3,3>(stream, x3a, I(29), I(30), x3b, B, nf, nf, 0, 7, H, W, nf, nf, 0, 7, H, W, 1, 1, 1, 1, 0);
  {
    int n = B * nf * HW;
    mean_d_k<<<(n + 255) / 256, 256, 0, stream>>>(x3b, xmean, B * nf, 7, HW);
  }
  launch_conv<MODE_CONV,3,1>(stream, xmean, I(31), I(32), tfbuf, B, nf, nf, 0, 1, H, W, nf, nf, 0, 1, H, W, 1, 1, 0, 0, 0); // no relu

  // ---- fuse + offset/mask head ----
  {
    int n = B * nf * HW;
    add_relu_k<<<(n + 255) / 256, 256, 0, stream>>>(tfull, tfbuf, fused, n);
  }
  launch_conv<MODE_CONV,3,1>(stream, fused, I(25), I(26), woutb, B, nf, nf, 0, 1, H, W, nf, nf, 0, 1, H, W, 1, 1, 0, 1, 0);
  launch_conv<MODE_CONV,3,1>(stream, woutb, I(33), I(34), om,    B, nf, nf, 0, 1, H, W, C*27, C*27, 0, 1, H, W, 1, 1, 0, 0, 0);

  // ---- deformable conv: sample, then 1x1 WMMA GEMM over (c,k)=63 ----
  {
    int n = B * C * 9 * HW;
    deform_sample_k<<<(n + 255) / 256, 256, 0, stream>>>(inp, om, sampled, B, C, H, W);
  }
  launch_conv<MODE_CONV,1,1>(stream, sampled, I(35), I(36), (float*)d_out,
                             B, C * 9, C * 9, 0, 1, H, W, 64, 64, 0, 1, H, W, 1, 0, 0, 1, 0);
}
